// FastFocalLoss_23759759082229
// MI455X (gfx1250) — compile-verified, matrix operand order checked by reading, not run on hardware
//
#include <hip/hip_runtime.h>

typedef float v4f __attribute__((ext_vector_type(4)));
typedef int   v8i __attribute__((ext_vector_type(8)));

#define BLOCK 256
#define WAVES (BLOCK / 32)

// ---------------------------------------------------------------------------
// Kernel 1: streaming partial reduction.
//  - b128 non-temporal loads (data is 402MB, read once: bypass-friendly TH)
//  - neg term every element, pos term only when the wave actually has a
//    positive (ballot check; ~12% of wave-iterations)
//  - count reduction across the wave via V_WMMA_I32_16X16X64_IU8 (B = ones)
//  - float sums via wave32 shfl_xor butterfly (fixed order -> deterministic)
//  - per-block partials written to ws (no float atomics)
// ---------------------------------------------------------------------------
__global__ __launch_bounds__(BLOCK) void focal_partial_kernel(
    const float* __restrict__ outp, const float* __restrict__ tgtp,
    long long n4,
    float* __restrict__ negP, float* __restrict__ posP, float* __restrict__ cntP)
{
    const long long tid    = (long long)blockIdx.x * BLOCK + threadIdx.x;
    const long long stride = (long long)gridDim.x * BLOCK;
    const v4f* o4 = (const v4f*)outp;
    const v4f* t4 = (const v4f*)tgtp;

    float    negS   = 0.0f;
    float    posS   = 0.0f;
    unsigned cntPk  = 0u;    // 4 packed u8 counters, one per float4 slot
    float    cntSp  = 0.0f;  // overflow spill (only for huge N)
    unsigned iters  = 0u;

    for (long long i = tid; i < n4; i += stride) {
        v4f o = __builtin_nontemporal_load(&o4[i]);
        v4f t = __builtin_nontemporal_load(&t4[i]);

        // negative term: log(1-o) * o^2 * (1-t)^4  (exactly 0 where t==1)
#pragma unroll
        for (int j = 0; j < 4; ++j) {
            float oo  = o[j];
            float omt = 1.0f - t[j];
            float g2  = omt * omt;
            negS += __logf(1.0f - oo) * (oo * oo) * (g2 * g2);
        }

        unsigned m0 = (t[0] == 1.0f) ? 1u : 0u;
        unsigned m1 = (t[1] == 1.0f) ? 1u : 0u;
        unsigned m2 = (t[2] == 1.0f) ? 1u : 0u;
        unsigned m3 = (t[3] == 1.0f) ? 1u : 0u;
        unsigned mp = m0 | (m1 << 8) | (m2 << 16) | (m3 << 24);

        if (__any(mp != 0u)) {  // rare positives: skip log(o) most of the time
#pragma unroll
            for (int j = 0; j < 4; ++j) {
                float oo  = o[j];
                float omo = 1.0f - oo;
                posS += (t[j] == 1.0f) ? __logf(oo) * omo * omo : 0.0f;
            }
            cntPk += mp;
        }

        if (((++iters) & 127u) == 0u) {  // keep u8 counters from saturating
            cntSp += (float)((cntPk & 0xFFu) + ((cntPk >> 8) & 0xFFu) +
                             ((cntPk >> 16) & 0xFFu) + ((cntPk >> 24) & 0xFFu));
            cntPk = 0u;
        }
    }

    // ---- whole-wave positive count with one IU8 WMMA -----------------------
    // A: each lane contributes 4 count bytes (rest zero). B: all-ones bytes.
    // D[m][n] = sum_k A[m][k]  => sum over all 16 rows == sum of ALL A bytes,
    // which is the exact 32-lane count, independent of the byte->(m,k) map.
    v8i a = {};
    a[0] = (int)cntPk;
    v8i b;
#pragma unroll
    for (int j = 0; j < 8; ++j) b[j] = 0x01010101;
    v8i c = {};
    v8i d = __builtin_amdgcn_wmma_i32_16x16x64_iu8(false, a, false, b, c, false, false);
    int s = d[0] + d[1] + d[2] + d[3] + d[4] + d[5] + d[6] + d[7];
    int waveCnt = s + __shfl_xor(s, 16, 32);   // rows 0..7 + rows 8..15

    // ---- wave32 butterfly for the float accumulators -----------------------
#pragma unroll
    for (int off = 16; off > 0; off >>= 1) {
        negS  += __shfl_xor(negS,  off, 32);
        posS  += __shfl_xor(posS,  off, 32);
        cntSp += __shfl_xor(cntSp, off, 32);
    }
    float cntW = (float)waveCnt + cntSp;

    // ---- cross-wave combine in LDS, one partial per block ------------------
    __shared__ float sNeg[WAVES], sPos[WAVES], sCnt[WAVES];
    const int wave = threadIdx.x >> 5;
    const int lane = threadIdx.x & 31;
    if (lane == 0) { sNeg[wave] = negS; sPos[wave] = posS; sCnt[wave] = cntW; }
    __syncthreads();
    if (threadIdx.x == 0) {
        float n = 0.0f, p = 0.0f, cc = 0.0f;
#pragma unroll
        for (int w = 0; w < WAVES; ++w) { n += sNeg[w]; p += sPos[w]; cc += sCnt[w]; }
        negP[blockIdx.x] = n;
        posP[blockIdx.x] = p;
        cntP[blockIdx.x] = cc;
    }
}

// ---------------------------------------------------------------------------
// Kernel 2: deterministic finalize (single block). Reduces per-block partials
// in a fixed order, folds any N%4 tail, applies the num_pos==0 select.
// ---------------------------------------------------------------------------
__global__ __launch_bounds__(BLOCK) void focal_final_kernel(
    const float* __restrict__ negP, const float* __restrict__ posP,
    const float* __restrict__ cntP, int nPart,
    const float* __restrict__ outp, const float* __restrict__ tgtp,
    long long tailStart, long long nTail,
    float* __restrict__ res)
{
    float n = 0.0f, p = 0.0f, c = 0.0f;
    for (int i = threadIdx.x; i < nPart; i += BLOCK) {
        n += negP[i]; p += posP[i]; c += cntP[i];
    }
#pragma unroll
    for (int off = 16; off > 0; off >>= 1) {
        n += __shfl_xor(n, off, 32);
        p += __shfl_xor(p, off, 32);
        c += __shfl_xor(c, off, 32);
    }
    __shared__ float sN[WAVES], sP[WAVES], sC[WAVES];
    const int wave = threadIdx.x >> 5;
    const int lane = threadIdx.x & 31;
    if (lane == 0) { sN[wave] = n; sP[wave] = p; sC[wave] = c; }
    __syncthreads();
    if (threadIdx.x == 0) {
        float tn = 0.0f, tp = 0.0f, tc = 0.0f;
#pragma unroll
        for (int w = 0; w < WAVES; ++w) { tn += sN[w]; tp += sP[w]; tc += sC[w]; }
        for (long long i = 0; i < nTail; ++i) {      // N%4 leftovers (0 here)
            float oo = outp[tailStart + i], tt = tgtp[tailStart + i];
            float omt = 1.0f - tt;
            float g2 = omt * omt;
            tn += __logf(1.0f - oo) * oo * oo * (g2 * g2);
            if (tt == 1.0f) { float omo = 1.0f - oo; tp += __logf(oo) * omo * omo; tc += 1.0f; }
        }
        *res = (tc == 0.0f) ? -tn : -((tp + tn) / tc);
    }
}

extern "C" void kernel_launch(void* const* d_in, const int* in_sizes, int n_in,
                              void* d_out, int out_size, void* d_ws, size_t ws_size,
                              hipStream_t stream) {
    const float* outp = (const float*)d_in[0];
    const float* tgtp = (const float*)d_in[1];
    const long long N         = (long long)in_sizes[0];
    const long long n4        = N >> 2;
    const long long tailStart = n4 << 2;
    const long long nTail     = N - tailStart;

    int blocks = 2048;                                   // 16K wave32s in flight
    const size_t perBlk = 3 * sizeof(float);
    if ((size_t)blocks * perBlk > ws_size) {
        blocks = (int)(ws_size / perBlk);
        if (blocks < 1) blocks = 1;
    }

    float* negP = (float*)d_ws;
    float* posP = negP + blocks;
    float* cntP = posP + blocks;

    focal_partial_kernel<<<blocks, BLOCK, 0, stream>>>(outp, tgtp, n4, negP, posP, cntP);
    focal_final_kernel<<<1, BLOCK, 0, stream>>>(negP, posP, cntP, blocks,
                                                outp, tgtp, tailStart, nTail,
                                                (float*)d_out);
}
